// EDUTEM_f_68521908240971
// MI455X (gfx1250) — matrix operand order repaired and checked by reading, not compiled, not run on hardware
//
#include <hip/hip_runtime.h>
#include <hip/hip_bf16.h>

typedef __attribute__((ext_vector_type(16))) _Float16 v16h;
typedef __attribute__((ext_vector_type(8)))  _Float16 v8h;
typedef __attribute__((ext_vector_type(8)))  float    v8f;

#define B_DIM 64
#define T_DIM 48
#define F_DIM 128
#define E_DIM 64
#define CD_DIM 32

// LDS strides (in _Float16 elements) with padding to dodge bank conflicts
#define S_C   72    // c / cw, [128][72]
#define S_T   136   // ct / attn / a / wt, [..][136]

// LDS byte offsets
#define OFF_CF16   0                         // 128*72*2  = 18432
#define OFF_CW16   (OFF_CF16 + 128*S_C*2)    // 18432
#define OFF_CT16   (OFF_CW16 + 128*S_C*2)    // 36864, ct: [64][136]
#define OFF_ATTN   (OFF_CT16 + 64*S_T*2)     // 54272, attn: [128][136]
#define OFF_A16    (OFF_ATTN + 128*S_T*2)    // 89088, a: [128][136]
#define OFF_WT16   (OFF_A16  + 128*S_T*2)    // 123904, wt: [32][136]
#define OFF_SCF32  (OFF_WT16 + 32*S_T*2)     // 132608, scores/exp [128][128] f32; reused as out staging
#define OFF_MLDS   (OFF_SCF32 + 128*128*4)   // 198144, m: [128] f32
#define OFF_BIAS   (OFF_MLDS + 128*4)        // 198656, bias: [128] f32
#define OFF_RSUM   (OFF_BIAS + 128*4)        // 199168, 1/rowsum: [128] f32
#define SMEM_BYTES (OFF_RSUM + 128*4)        // 199680

// ---- WMMA fragment loaders from LDS (layouts per CDNA5 ISA 7.12.2) ----

// A fragment (16x32 f16): lane (l&15) = row, half h = l>>4.
// v[0..7]  <- K = k0 + h*8 + (0..7)
// v[8..15] <- K = k0 + 16 + h*8 + (0..7)
__device__ __forceinline__ v16h load_a_frag(const _Float16* m, int row0, int stride, int k0) {
    int lane = threadIdx.x & 31;
    int r = lane & 15, h = lane >> 4;
    const _Float16* p = m + (row0 + r) * stride + k0 + h * 8;
    v8h lo = *(const v8h*)p;
    v8h hi = *(const v8h*)(p + 16);
    v16h out;
#pragma unroll
    for (int i = 0; i < 8; ++i) { out[i] = lo[i]; out[8 + i] = hi[i]; }
    return out;
}

// B fragment (32x16 f16): lane (l&15) = column N, half h = l>>4.
// v[0..7] <- K = k0 + h*8 + (0..7), v[8..15] <- K = k0 + 16 + h*8 + (0..7), for fixed N.
// Source matrix is stored "N-major": mat[n][k].
__device__ __forceinline__ v16h load_b_frag(const _Float16* m, int n0, int stride, int k0) {
    int lane = threadIdx.x & 31;
    int n = lane & 15, h = lane >> 4;
    const _Float16* p = m + (n0 + n) * stride + k0 + h * 8;
    v8h lo = *(const v8h*)p;
    v8h hi = *(const v8h*)(p + 16);
    v16h out;
#pragma unroll
    for (int i = 0; i < 8; ++i) { out[i] = lo[i]; out[8 + i] = hi[i]; }
    return out;
}

__device__ __forceinline__ v8f wmma_f16(v16h a, v16h b, v8f c) {
    return __builtin_amdgcn_wmma_f32_16x16x32_f16(
        /*neg_a=*/false, a, /*neg_b=*/false, b,
        /*c_mod=*/(short)0, c, /*reuse_a=*/false, /*reuse_b=*/false);
}

__global__ __launch_bounds__(256, 1)
void edutem_fused_kernel(const float* __restrict__ input_x,
                         const int*   __restrict__ mask,
                         const float* __restrict__ embed0,
                         const float* __restrict__ embed1,
                         const float* __restrict__ embed_missing,
                         const float* __restrict__ att_w,
                         const float* __restrict__ att_b,
                         const float* __restrict__ comp_w,
                         float*       __restrict__ out) {
    extern __shared__ char smem[];
    _Float16* cf16 = (_Float16*)(smem + OFF_CF16);  // c,  [f][e] stride S_C
    _Float16* cw16 = (_Float16*)(smem + OFF_CW16);  // c*w,[f][e] stride S_C
    _Float16* ct16 = (_Float16*)(smem + OFF_CT16);  // c,  [e][f] stride S_T
    _Float16* at16 = (_Float16*)(smem + OFF_ATTN);  // attn [f][f] stride S_T
    _Float16* a16  = (_Float16*)(smem + OFF_A16);   // relu concat [f][2e] stride S_T
    _Float16* wt16 = (_Float16*)(smem + OFF_WT16);  // comp_w^T [d][k] stride S_T
    float*    scf  = (float*)(smem + OFF_SCF32);    // exp scores / out staging
    float*    mLds = (float*)(smem + OFF_MLDS);
    float*    bLds = (float*)(smem + OFF_BIAS);
    float*    rInv = (float*)(smem + OFF_RSUM);

    const int tid  = threadIdx.x;
    const int wave = tid >> 5;
    const int lane = tid & 31;
    const int nlo  = lane & 15;
    const int half = lane >> 4;
    const int bt   = blockIdx.x;          // 0 .. B*T-1
    const int b    = bt / T_DIM;

    // L2 prefetch of the shared tables (global_prefetch_b8)
    {
        const char* p0 = (const char*)embed0        + (size_t)tid * 128;
        const char* p1 = (const char*)embed1        + (size_t)tid * 128;
        const char* p2 = (const char*)embed_missing + (size_t)tid * 128;
        const char* p3 = (const char*)att_w         + (size_t)tid * 128;
        __builtin_prefetch(p0, 0, 1);
        __builtin_prefetch(p1, 0, 1);
        __builtin_prefetch(p2, 0, 1);
        __builtin_prefetch(p3, 0, 1);
        if (tid < 128) __builtin_prefetch((const char*)comp_w + (size_t)tid * 128, 0, 1);
    }

    // ---------------- Phase 0: per-batch feature mask + bias -------------
    if (tid < F_DIM) {
        int f = tid;
        const int* mp = mask + (size_t)b * T_DIM * F_DIM + f;
        int s = 0;
#pragma unroll 4
        for (int t = 0; t < T_DIM; ++t) s += mp[t * F_DIM];
        mLds[f] = (s != 0) ? 1.0f : 0.0f;
        bLds[f] = att_b[f];
    }
    __syncthreads();

    // ---------------- Phase 1: build c in all needed layouts -------------
    const float* xrow = input_x + (size_t)bt * F_DIM;
#pragma unroll
    for (int idx = tid; idx < F_DIM * E_DIM; idx += 256) {
        int f = idx >> 6, e = idx & 63;
        float x = xrow[f];
        float m = mLds[f];
        // CLIP_MIN=0, CLIP_MAX=1 -> c = x*e0 + (m-x)*e1 + (1-m)*em
        float c = x * embed0[idx] + (m - x) * embed1[idx]
                + (1.0f - m) * embed_missing[idx];
        _Float16 ch = (_Float16)c;
        cf16[f * S_C + e] = ch;
        cw16[f * S_C + e] = (_Float16)(c * att_w[idx]);
        ct16[e * S_T + f] = ch;
        a16 [f * S_T + e] = (_Float16)fmaxf(c, 0.0f);   // relu(c) half of concat
    }
    // compress_w [128][32] -> transposed f16 [d][k]
    for (int idx = tid; idx < F_DIM * CD_DIM; idx += 256) {
        int k = idx >> 5, d = idx & 31;
        wt16[d * S_T + k] = (_Float16)comp_w[idx];
    }
    __syncthreads();

    // ------- Phase 2: scores = (c.w) @ c^T  (M=128,N=128,K=64) ----------
    // wave w owns M-tile w, all 8 N-tiles; A frags + bias are loop-invariant.
    {
        const int mt = wave;
        v16h a0 = load_a_frag(cw16, 16 * mt, S_C, 0);
        v16h a1 = load_a_frag(cw16, 16 * mt, S_C, 32);
        float bias[8], psum[8];
#pragma unroll
        for (int i = 0; i < 8; ++i) {
            bias[i] = bLds[16 * mt + i + 8 * half];
            psum[i] = 0.0f;
        }
#pragma unroll
        for (int nt = 0; nt < 8; ++nt) {
            v8f acc = {};
            acc = wmma_f16(a0, load_b_frag(cf16, 16 * nt, S_C, 0),  acc);
            acc = wmma_f16(a1, load_b_frag(cf16, 16 * nt, S_C, 32), acc);
            int n = nlo + 16 * nt;
#pragma unroll
            for (int i = 0; i < 8; ++i) {
                int mr = 16 * mt + i + 8 * half;
                float s = fminf(5.0f, fmaxf(-5.0f, acc[i] + bias[i]));
                float ex = (mr == n) ? 0.0f : __expf(s);   // off-diagonal mask
                psum[i] += ex;
                scf[mr * F_DIM + n] = ex;
            }
        }
        // reduce row sums across the 16 lanes of each half; store reciprocal
#pragma unroll
        for (int i = 0; i < 8; ++i) {
            float s = psum[i];
            s += __shfl_xor(s, 8, 32);
            s += __shfl_xor(s, 4, 32);
            s += __shfl_xor(s, 2, 32);
            s += __shfl_xor(s, 1, 32);
            if (nlo == 0) rInv[16 * mt + i + 8 * half] = 1.0f / (s + 1e-8f);
        }
    }
    __syncthreads();

    // ---------------- Phase 3: attn = exp * (1/rowsum) -> f16 ------------
#pragma unroll
    for (int idx = tid; idx < F_DIM * F_DIM; idx += 256) {
        int r = idx >> 7, n = idx & 127;
        at16[r * S_T + n] = (_Float16)(scf[idx] * rInv[r]);
    }
    __syncthreads();

    // ------- Phase 4: agg = c * (attn @ c)  (M=128,N=64,K=128) ----------
    // wave w owns M-tile w, 4 N-tiles; A frags loop-invariant.
    {
        const int mt = wave;
        v16h af[4];
#pragma unroll
        for (int k = 0; k < 4; ++k) af[k] = load_a_frag(at16, 16 * mt, S_T, 32 * k);
#pragma unroll
        for (int nt = 0; nt < 4; ++nt) {
            v8f acc = {};
#pragma unroll
            for (int k = 0; k < 4; ++k)
                acc = wmma_f16(af[k], load_b_frag(ct16, 16 * nt, S_T, 32 * k), acc);
            int e = nlo + 16 * nt;
#pragma unroll
            for (int i = 0; i < 8; ++i) {
                int f = 16 * mt + i + 8 * half;
                float cv  = (float)cf16[f * S_C + e];
                a16[f * S_T + E_DIM + e] = (_Float16)fmaxf(cv * acc[i], 0.0f);
            }
        }
    }
    __syncthreads();

    // ------- Phase 5: out = a @ comp_w  (M=128,N=32,K=128) --------------
    // wave w owns M-tile w, 2 N-tiles; stage result tile in LDS (reuse scf).
    {
        const int mt = wave;
        v16h af[4];
#pragma unroll
        for (int k = 0; k < 4; ++k) af[k] = load_a_frag(a16, 16 * mt, S_T, 32 * k);
#pragma unroll
        for (int nt = 0; nt < 2; ++nt) {
            v8f acc = {};
#pragma unroll
            for (int k = 0; k < 4; ++k)
                acc = wmma_f16(af[k], load_b_frag(wt16, 16 * nt, S_T, 32 * k), acc);
            int d = nlo + 16 * nt;
#pragma unroll
            for (int i = 0; i < 8; ++i) {
                int f = 16 * mt + i + 8 * half;
                scf[f * CD_DIM + d] = acc[i];
            }
        }
    }
    __syncthreads();

    // coalesced float4 writeback of the 128x32 f32 tile
    {
        float4* dst = (float4*)(out + (size_t)bt * F_DIM * CD_DIM);
        const float4* src = (const float4*)scf;
#pragma unroll
        for (int idx = tid; idx < (F_DIM * CD_DIM) / 4; idx += 256)
            dst[idx] = src[idx];
    }
}

extern "C" void kernel_launch(void* const* d_in, const int* in_sizes, int n_in,
                              void* d_out, int out_size, void* d_ws, size_t ws_size,
                              hipStream_t stream) {
    (void)in_sizes; (void)n_in; (void)out_size; (void)d_ws; (void)ws_size;
    const float* input_x       = (const float*)d_in[0];
    const int*   mask          = (const int*)  d_in[1];
    const float* embed0        = (const float*)d_in[2];
    const float* embed1        = (const float*)d_in[3];
    const float* embed_missing = (const float*)d_in[4];
    const float* att_w         = (const float*)d_in[5];
    const float* att_b         = (const float*)d_in[6];
    const float* comp_w        = (const float*)d_in[7];
    float* out = (float*)d_out;

    // > 64 KB dynamic LDS: raise the limit (deterministic, host-side, capture-safe)
    hipFuncSetAttribute((const void*)edutem_fused_kernel,
                        hipFuncAttributeMaxDynamicSharedMemorySize, SMEM_BYTES);

    dim3 grid(B_DIM * T_DIM);
    dim3 block(256);
    edutem_fused_kernel<<<grid, block, SMEM_BYTES, stream>>>(
        input_x, mask, embed0, embed1, embed_missing, att_w, att_b, comp_w, out);
}
